// QMixtralAttention_58256936403175
// MI455X (gfx1250) — matrix operand order repaired and checked by reading, not compiled, not run on hardware
//
#include <hip/hip_runtime.h>

// ---------------------------------------------------------------------------
// Types for CDNA5 WMMA (wave32): A/B = 16 bf16 elems/lane, C/D = 8 f32/lane
// ---------------------------------------------------------------------------
typedef __bf16 bf16x16 __attribute__((ext_vector_type(16)));
typedef __bf16 bf16x8  __attribute__((ext_vector_type(8)));
typedef float  v8f     __attribute__((ext_vector_type(8)));
typedef unsigned int u32x4 __attribute__((ext_vector_type(4)));
typedef int i32x4 __attribute__((ext_vector_type(4)));
typedef int i32x8 __attribute__((ext_vector_type(8)));

#define S_LEN 2048
#define HID   4096
#define NH    32
#define NKV   8
#define HD    128
#define SOFT_SCALE 0.08838834764831845f  // 1/sqrt(128)

__device__ __forceinline__ v8f wmma_bf16(bf16x16 a, bf16x16 b, v8f c) {
  // D = A(16x32) * B(32x16) + C, f32 accumulate
  return __builtin_amdgcn_wmma_f32_16x16x32_bf16(
      /*neg_a=*/false, a, /*neg_b=*/false, b,
      /*c_mod=*/(short)0, c, /*reuse_a=*/false, /*reuse_b=*/false);
}

// A-matrix fragment: 16x32 bf16, row-major source, lda elems.
// ISA layout: lane L holds row M = m0 + L%16; elems 0..7 -> K = (L/16)*8+0..7,
// elems 8..15 -> K = 16 + (L/16)*8 + 0..7  (two contiguous 16B runs).
__device__ __forceinline__ bf16x16 load_A_frag(const __bf16* A, int lda,
                                               int m0, int k0) {
  int lane = threadIdx.x & 31;
  int row  = m0 + (lane & 15);
  int half = lane >> 4;
  const __bf16* p = A + (size_t)row * lda + k0 + half * 8;
  bf16x8 lo = *(const bf16x8*)p;
  bf16x8 hi = *(const bf16x8*)(p + 16);
  bf16x16 r;
#pragma unroll
  for (int i = 0; i < 8; ++i) { r[i] = lo[i]; r[i + 8] = hi[i]; }
  return r;
}

// B-matrix fragment (B = W^T), W stored row-major [N][K], ldw elems.
// ISA layout: lane L holds col N = n0 + L%16; elems 0..7 -> K = (L/16)*8+0..7,
// elems 8..15 -> K = 16 + (L/16)*8 + 0..7  (two contiguous 16B runs).
__device__ __forceinline__ bf16x16 load_B_frag(const __bf16* W, int ldw,
                                               int n0, int k0) {
  int lane = threadIdx.x & 31;
  int n    = n0 + (lane & 15);
  int kh   = lane >> 4;
  const __bf16* p = W + (size_t)n * ldw + k0 + kh * 8;
  bf16x8 lo = *(const bf16x8*)p;
  bf16x8 hi = *(const bf16x8*)(p + 16);
  bf16x16 r;
#pragma unroll
  for (int i = 0; i < 8; ++i) { r[i] = lo[i]; r[i + 8] = hi[i]; }
  return r;
}

// ---------------------------------------------------------------------------
// Tensor Data Mover: 2D tile (tile_rows x tile_k bf16) global -> LDS.
// D# layout per CDNA5 ISA:
//   group0: [1:0]=count=1, [63:32]=lds_addr, [120:64]=global_addr, [127:126]=2
//   group1: [17:16]=data_size(1=2B), [79:48]=tensor_dim0, [111:80]=tensor_dim1,
//           [127:112]=tile_dim0, [143:128]=tile_dim1, [207:160]=dim0_stride
// Tracked with TENSORcnt; caller waits s_wait_tensorcnt + barrier.
// ---------------------------------------------------------------------------
__device__ __forceinline__ void tdm_load_2d(unsigned lds_off,
                                            const __bf16* gptr,
                                            unsigned k_total,
                                            unsigned rows_total,
                                            unsigned tile_k,
                                            unsigned tile_rows) {
  unsigned long long ga = (unsigned long long)(uintptr_t)gptr;
  u32x4 g0;
  g0[0] = 1u;                                   // count=1, user descriptor
  g0[1] = lds_off;                              // LDS byte address
  g0[2] = (unsigned)(ga & 0xFFFFFFFFu);         // global_addr[31:0]
  g0[3] = (unsigned)((ga >> 32) & 0x01FFFFFFu)  // global_addr[56:32]
          | 0x80000000u;                        // type = 2 ("image")
  i32x8 g1;
  g1[0] = (int)(1u << 16);                                      // data_size=2B
  g1[1] = (int)((k_total & 0xFFFFu) << 16);                     // dim0 lo16
  g1[2] = (int)((k_total >> 16) | ((rows_total & 0xFFFFu) << 16));
  g1[3] = (int)((rows_total >> 16) | (tile_k << 16));           // tile_dim0
  g1[4] = (int)tile_rows;                                       // tile_dim1
  g1[5] = (int)k_total;                                         // dim0_stride
  g1[6] = 0;
  g1[7] = 0;
  i32x4 z4 = {0, 0, 0, 0};
#if defined(__clang_major__) && (__clang_major__ >= 23)
  i32x8 z8 = {0, 0, 0, 0, 0, 0, 0, 0};
  __builtin_amdgcn_tensor_load_to_lds(g0, g1, z4, z4, z8, 0);
#else
  __builtin_amdgcn_tensor_load_to_lds(g0, g1, z4, z4, 0);
#endif
}

// ---------------------------------------------------------------------------
// TDM-staged bf16 GEMM: C[M][N] (f32) = A[M][K] (bf16) * W[N][K]^T (bf16)
// Block = 256 thr = 8 waves (2x4); block tile 128(M) x 128(N); K-stage 64,
// double-buffered LDS fed by tensor_load_to_lds. Wave tile 64x32 = 4x2 frags.
// ---------------------------------------------------------------------------
__global__ void __launch_bounds__(256)
gemm_bf16_tdm(const __bf16* __restrict__ A, const __bf16* __restrict__ W,
              float* __restrict__ C, int M, int N, int K) {
  __shared__ __bf16 Abuf[2][128 * 64];   // 16 KB each
  __shared__ __bf16 Bbuf[2][128 * 64];

  int lane = threadIdx.x & 31;
  int wid  = threadIdx.x >> 5;
  int wm   = wid >> 2;       // 0..1 -> 64-row slabs
  int wn   = wid & 3;        // 0..3 -> 32-col slabs
  int m0   = blockIdx.y * 128;
  int n0   = blockIdx.x * 128;

  v8f acc[4][2] = {};

  // prologue: stage 0
  if (wid == 0) {
    tdm_load_2d((unsigned)(uintptr_t)&Abuf[0][0], A + (size_t)m0 * K,
                (unsigned)K, (unsigned)M, 64u, 128u);
    tdm_load_2d((unsigned)(uintptr_t)&Bbuf[0][0], W + (size_t)n0 * K,
                (unsigned)K, (unsigned)N, 64u, 128u);
    __builtin_amdgcn_s_wait_tensorcnt(0);
  }
  __syncthreads();

  int cur = 0;
  for (int k = 0; k < K; k += 64) {
    int kn = k + 64;
    if (kn < K && wid == 0) {
      tdm_load_2d((unsigned)(uintptr_t)&Abuf[cur ^ 1][0],
                  A + (size_t)m0 * K + kn, (unsigned)K, (unsigned)M, 64u, 128u);
      tdm_load_2d((unsigned)(uintptr_t)&Bbuf[cur ^ 1][0],
                  W + (size_t)n0 * K + kn, (unsigned)K, (unsigned)N, 64u, 128u);
    }

#pragma unroll
    for (int ks = 0; ks < 64; ks += 32) {
      bf16x16 af[4], bfr[2];
#pragma unroll
      for (int i = 0; i < 4; ++i)
        af[i] = load_A_frag(&Abuf[cur][0], 64, wm * 64 + i * 16, ks);
#pragma unroll
      for (int j = 0; j < 2; ++j)
        bfr[j] = load_B_frag(&Bbuf[cur][0], 64, wn * 32 + j * 16, ks);
#pragma unroll
      for (int i = 0; i < 4; ++i)
#pragma unroll
        for (int j = 0; j < 2; ++j)
          acc[i][j] = wmma_bf16(af[i], bfr[j], acc[i][j]);
    }

    if (kn < K) {
      if (wid == 0) __builtin_amdgcn_s_wait_tensorcnt(0);
      __syncthreads();
      cur ^= 1;
    }
  }

  int half = lane >> 4;
#pragma unroll
  for (int i = 0; i < 4; ++i)
#pragma unroll
    for (int j = 0; j < 2; ++j) {
      int col = n0 + wn * 32 + j * 16 + (lane & 15);
#pragma unroll
      for (int v = 0; v < 8; ++v) {
        int row = m0 + wm * 64 + i * 16 + v + 8 * half;
        C[(size_t)row * N + col] = acc[i][j][v];
      }
    }
}

// ---------------------------------------------------------------------------
// Fallback direct-load GEMM (kept compiled; not launched)
// ---------------------------------------------------------------------------
__global__ void __launch_bounds__(256)
gemm_bf16_nt(const __bf16* __restrict__ A, const __bf16* __restrict__ W,
             float* __restrict__ C, int M, int N, int K) {
  int lane = threadIdx.x & 31;
  int wid  = threadIdx.x >> 5;
  int m0   = blockIdx.y * 64 + (wid >> 2) * 32;
  int n0   = blockIdx.x * 128 + (wid & 3) * 32;
  (void)M;
  v8f acc[2][2] = {};
  for (int k = 0; k < K; k += 32) {
    bf16x16 a0 = load_A_frag(A, K, m0, k);
    bf16x16 a1 = load_A_frag(A, K, m0 + 16, k);
    bf16x16 b0 = load_B_frag(W, K, n0, k);
    bf16x16 b1 = load_B_frag(W, K, n0 + 16, k);
    acc[0][0] = wmma_bf16(a0, b0, acc[0][0]);
    acc[0][1] = wmma_bf16(a0, b1, acc[0][1]);
    acc[1][0] = wmma_bf16(a1, b0, acc[1][0]);
    acc[1][1] = wmma_bf16(a1, b1, acc[1][1]);
  }
  int half = lane >> 4;
#pragma unroll
  for (int i = 0; i < 2; ++i)
#pragma unroll
    for (int j = 0; j < 2; ++j) {
      int col = n0 + j * 16 + (lane & 15);
#pragma unroll
      for (int v = 0; v < 8; ++v)
        C[(size_t)(m0 + i * 16 + v + 8 * half) * N + col] = acc[i][j][v];
    }
}

// ---------------------------------------------------------------------------
// Elementwise f32 -> bf16 conversion
// ---------------------------------------------------------------------------
__global__ void __launch_bounds__(256)
f32_to_bf16_kernel(const float* __restrict__ in, __bf16* __restrict__ out,
                   long n) {
  long i = (long)blockIdx.x * 256 + threadIdx.x;
  if (i < n) out[i] = (__bf16)in[i];
}

// ---------------------------------------------------------------------------
// RoPE + repack to per-head layout: x[S][nh*128] (f32) -> out[nh][S][128] bf16
// position_ids = arange(S); theta = 1e6
// ---------------------------------------------------------------------------
__global__ void __launch_bounds__(256)
rope_to_heads(const float* __restrict__ x, __bf16* __restrict__ out, int nh) {
  int idx  = blockIdx.x * 256 + threadIdx.x;  // over S * nh * 64
  int d    = idx & 63;
  int rest = idx >> 6;
  int h    = rest % nh;
  int s    = rest / nh;
  float inv = __expf((float)d * -0.21586735246819178f);  // ln(1e6)/64
  float ang = (float)s * inv;
  float c = __cosf(ang), sn = __sinf(ang);
  const float* row = x + (size_t)s * (nh * HD) + h * HD;
  float x1 = row[d], x2 = row[d + 64];
  __bf16* o = out + ((size_t)h * S_LEN + s) * HD;
  o[d]      = (__bf16)(x1 * c - x2 * sn);
  o[d + 64] = (__bf16)(x2 * c + x1 * sn);
}

// V repack transposed: v[S][NKV*128] (f32) -> vt[NKV][128][S] (bf16)
__global__ void __launch_bounds__(256)
v_transpose_bf16(const float* __restrict__ v, __bf16* __restrict__ vt) {
  int idx = blockIdx.x * 256 + threadIdx.x;  // over S*NKV*HD
  int d   = idx & 127;
  int kh  = (idx >> 7) & 7;
  int s   = idx >> 10;
  vt[((size_t)kh * HD + d) * S_LEN + s] =
      (__bf16)v[(size_t)s * (NKV * HD) + kh * HD + d];
}

// ---------------------------------------------------------------------------
// Causal GQA flash attention.
//   q: [NH][S][HD] bf16 (RoPE'd), k: [NKV][S][HD] bf16 (RoPE'd),
//   vt: [NKV][HD][S] bf16, out: [S][NH*HD] bf16
// Block = 128 threads = 4 waves; each wave owns 16 query rows (online
// softmax is wave-private). Grid = (S/64, NH).
// ---------------------------------------------------------------------------
__global__ void __launch_bounds__(128)
flash_attn_bf16(const __bf16* __restrict__ q, const __bf16* __restrict__ k,
                const __bf16* __restrict__ vt, __bf16* __restrict__ out) {
  __shared__ __bf16 plds[4][16 * 32];  // wave-private 16x32 P staging

  int lane = threadIdx.x & 31;
  int wid  = threadIdx.x >> 5;
  int half = lane >> 4;
  int cn   = lane & 15;

  int h  = blockIdx.y;
  int kh = h >> 2;  // N_REP = 4
  int q0 = blockIdx.x * 64 + wid * 16;

  const __bf16* qh  = q + (size_t)h * S_LEN * HD;
  const __bf16* kt  = k + (size_t)kh * S_LEN * HD;
  const __bf16* vtp = vt + (size_t)kh * HD * S_LEN;
  __bf16* pl = &plds[wid][0];

  bf16x16 qf[4];
#pragma unroll
  for (int t = 0; t < 4; ++t) qf[t] = load_A_frag(qh, HD, q0, t * 32);

  v8f o[8] = {};
  float mrun[8], lrun[8];
#pragma unroll
  for (int v = 0; v < 8; ++v) { mrun[v] = -1e30f; lrun[v] = 0.0f; }

  for (int kb = 0; kb < q0 + 16; kb += 32) {
    // ---- S = Q K^T / sqrt(HD) for 32 keys (two 16-col fragments) ----
    v8f s0 = {}, s1 = {};
#pragma unroll
    for (int t = 0; t < 4; ++t) {
      bf16x16 b0 = load_B_frag(kt, HD, kb, t * 32);
      bf16x16 b1 = load_B_frag(kt, HD, kb + 16, t * 32);
      s0 = wmma_bf16(qf[t], b0, s0);
      s1 = wmma_bf16(qf[t], b1, s1);
    }
#pragma unroll
    for (int v = 0; v < 8; ++v) { s0[v] *= SOFT_SCALE; s1[v] *= SOFT_SCALE; }

    // ---- causal mask (diagonal blocks only; wave-uniform condition) ----
    if (kb + 31 > q0) {
#pragma unroll
      for (int v = 0; v < 8; ++v) {
        int row = q0 + v + 8 * half;
        if (kb + cn > row)      s0[v] = -1e9f;
        if (kb + 16 + cn > row) s1[v] = -1e9f;
      }
    }

    // ---- online softmax (rows live in 16-lane halves; butterfly reduce) ----
    float corr[8];
#pragma unroll
    for (int v = 0; v < 8; ++v) {
      float mx = fmaxf(s0[v], s1[v]);
#pragma unroll
      for (int off = 8; off; off >>= 1) mx = fmaxf(mx, __shfl_xor(mx, off, 32));
      float mnew = fmaxf(mrun[v], mx);
      float p0 = __expf(s0[v] - mnew);
      float p1 = __expf(s1[v] - mnew);
      float rs = p0 + p1;
#pragma unroll
      for (int off = 8; off; off >>= 1) rs += __shfl_xor(rs, off, 32);
      float cr = __expf(mrun[v] - mnew);
      lrun[v] = lrun[v] * cr + rs;
      mrun[v] = mnew;
      corr[v] = cr;
      s0[v] = p0;
      s1[v] = p1;
    }
#pragma unroll
    for (int j = 0; j < 8; ++j)
#pragma unroll
      for (int v = 0; v < 8; ++v) o[j][v] *= corr[v];

    // ---- P (C-layout regs) -> LDS -> A-layout fragment ----
#pragma unroll
    for (int v = 0; v < 8; ++v) {
      int row = v + 8 * half;
      pl[row * 32 + cn]      = (__bf16)s0[v];
      pl[row * 32 + 16 + cn] = (__bf16)s1[v];
    }
    asm volatile("s_wait_dscnt 0" ::: "memory");
    bf16x8 lo = *(const bf16x8*)&pl[(lane & 15) * 32 + half * 8];
    bf16x8 hi = *(const bf16x8*)&pl[(lane & 15) * 32 + 16 + half * 8];
    bf16x16 pa;
#pragma unroll
    for (int i = 0; i < 8; ++i) { pa[i] = lo[i]; pa[i + 8] = hi[i]; }

    // ---- O += P(16x32) * V(32x128), V supplied transposed ----
#pragma unroll
    for (int j = 0; j < 8; ++j) {
      bf16x16 bv = load_B_frag(vtp, S_LEN, j * 16, kb);
      o[j] = wmma_bf16(pa, bv, o[j]);
    }
  }

  // ---- epilogue: O / l, pack to [S][NH*HD] bf16 for the Wo GEMM ----
#pragma unroll
  for (int j = 0; j < 8; ++j) {
    int col = h * HD + j * 16 + cn;
#pragma unroll
    for (int v = 0; v < 8; ++v) {
      int srow = q0 + v + 8 * half;
      out[(size_t)srow * (NH * HD) + col] = (__bf16)(o[j][v] / lrun[v]);
    }
  }
}

// ---------------------------------------------------------------------------
// Launch: hs->bf16, W->bf16, QKV GEMMs (TDM), RoPE/repack, flash attn, Wo GEMM
// ---------------------------------------------------------------------------
extern "C" void kernel_launch(void* const* d_in, const int* in_sizes, int n_in,
                              void* d_out, int out_size, void* d_ws,
                              size_t ws_size, hipStream_t stream) {
  (void)in_sizes; (void)n_in; (void)out_size; (void)ws_size;

  const float* hs = (const float*)d_in[0];   // [S, HID]
  // d_in[1] attention_mask (causal, rebuilt in-kernel), d_in[2] position_ids
  const float* Wq = (const float*)d_in[3];   // [NH*HD, HID]
  const float* Wk = (const float*)d_in[4];   // [NKV*HD, HID]
  const float* Wv = (const float*)d_in[5];   // [NKV*HD, HID]
  const float* Wo = (const float*)d_in[6];   // [HID, NH*HD]
  float* outp = (float*)d_out;               // [S, HID]

  char* ws = (char*)d_ws;
  size_t off = 0;
  auto alloc = [&](size_t bytes) {
    size_t r = off;
    off += (bytes + 255) & ~(size_t)255;
    return r;
  };
  __bf16* hs_bf = (__bf16*)(ws + alloc((size_t)S_LEN * HID * 2));
  __bf16* wq_bf = (__bf16*)(ws + alloc((size_t)HID * HID * 2));
  __bf16* wk_bf = (__bf16*)(ws + alloc((size_t)NKV * HD * HID * 2));
  __bf16* wv_bf = (__bf16*)(ws + alloc((size_t)NKV * HD * HID * 2));
  float*  q_f   = (float*)(ws + alloc((size_t)S_LEN * NH * HD * 4));
  float*  k_f   = (float*)(ws + alloc((size_t)S_LEN * NKV * HD * 4));
  float*  v_f   = (float*)(ws + alloc((size_t)S_LEN * NKV * HD * 4));
  __bf16* q_bf  = (__bf16*)(ws + alloc((size_t)NH * S_LEN * HD * 2));
  __bf16* k_bf  = (__bf16*)(ws + alloc((size_t)NKV * S_LEN * HD * 2));
  __bf16* vt_bf = (__bf16*)(ws + alloc((size_t)NKV * HD * S_LEN * 2));
  // Dead-buffer reuse (stream-ordered):
  __bf16* attn_bf = (__bf16*)q_f;   // q_f dead after RoPE
  __bf16* wo_bf   = wq_bf;          // wq_bf dead after Q projection

  // 1) conversions to bf16
  f32_to_bf16_kernel<<<(S_LEN * HID) / 256, 256, 0, stream>>>(
      hs, hs_bf, (long)S_LEN * HID);
  f32_to_bf16_kernel<<<((size_t)HID * HID) / 256, 256, 0, stream>>>(
      Wq, wq_bf, (long)HID * HID);
  f32_to_bf16_kernel<<<((size_t)NKV * HD * HID) / 256, 256, 0, stream>>>(
      Wk, wk_bf, (long)NKV * HD * HID);
  f32_to_bf16_kernel<<<((size_t)NKV * HD * HID) / 256, 256, 0, stream>>>(
      Wv, wv_bf, (long)NKV * HD * HID);

  // 2) QKV projections (C = hs * W^T), TDM-staged
  gemm_bf16_tdm<<<dim3(HID / 128, S_LEN / 128), 256, 0, stream>>>(
      hs_bf, wq_bf, q_f, S_LEN, HID, HID);
  gemm_bf16_tdm<<<dim3((NKV * HD) / 128, S_LEN / 128), 256, 0, stream>>>(
      hs_bf, wk_bf, k_f, S_LEN, NKV * HD, HID);
  gemm_bf16_tdm<<<dim3((NKV * HD) / 128, S_LEN / 128), 256, 0, stream>>>(
      hs_bf, wv_bf, v_f, S_LEN, NKV * HD, HID);

  // 3) RoPE + per-head bf16 repack; V transposed repack
  rope_to_heads<<<(S_LEN * NH * 64) / 256, 256, 0, stream>>>(q_f, q_bf, NH);
  rope_to_heads<<<(S_LEN * NKV * 64) / 256, 256, 0, stream>>>(k_f, k_bf, NKV);
  v_transpose_bf16<<<(S_LEN * NKV * HD) / 256, 256, 0, stream>>>(v_f, vt_bf);

  // 4) causal GQA flash attention
  flash_attn_bf16<<<dim3(S_LEN / 64, NH), 128, 0, stream>>>(q_bf, k_bf, vt_bf,
                                                            attn_bf);

  // 5) output projection
  f32_to_bf16_kernel<<<((size_t)HID * HID) / 256, 256, 0, stream>>>(
      Wo, wo_bf, (long)HID * HID);
  gemm_bf16_tdm<<<dim3(HID / 128, S_LEN / 128), 256, 0, stream>>>(
      attn_bf, wo_bf, outp, S_LEN, HID, HID);
}
